// CrossAttention_3942779978031
// MI455X (gfx1250) — matrix operand order repaired and checked by reading, not compiled, not run on hardware
//
#include <hip/hip_runtime.h>
#include <hip/hip_bf16.h>

#define BB   4
#define LQ   4096
#define LK   4096
#define DIN  256
#define DOUT 128

typedef __attribute__((ext_vector_type(16))) __bf16 v16bf;
typedef __attribute__((ext_vector_type(8)))  __bf16 v8bf;
typedef __attribute__((ext_vector_type(8)))  float  v8f;

static __device__ inline v16bf cat8(v8bf lo, v8bf hi) {
  return __builtin_shufflevector(lo, hi, 0,1,2,3,4,5,6,7,8,9,10,11,12,13,14,15);
}

static __device__ inline v8f wmma_bf16(v16bf a, v16bf b, v8f c) {
  // D(f32 16x16) = A(bf16 16x32) * B(bf16 32x16) + C
  return __builtin_amdgcn_wmma_f32_16x16x32_bf16(false, a, false, b, (short)0, c,
                                                 false, false);
}

// ---------------------------------------------------------------------------
// Kernel 1: fused Q/K/V projection, fp32 -> bf16 workspace.
// grid = (16 tile-groups, 3 projections). Each block stages its weight matrix
// (256x128) once into LDS in B-fragment order (bf16, 64 KB), then 4 waves each
// run 16 row-tiles of WMMA against it. V is written transposed [b][d][k].
// ---------------------------------------------------------------------------
__global__ __launch_bounds__(128)
void proj_qkv(const float* __restrict__ x, const float* __restrict__ cond,
              const float* __restrict__ Wq, const float* __restrict__ bq,
              const float* __restrict__ Wk, const float* __restrict__ bk,
              const float* __restrict__ Wv, const float* __restrict__ bv,
              __bf16* __restrict__ q_ws, __bf16* __restrict__ k_ws,
              __bf16* __restrict__ vt_ws)
{
  // [kchunk 0..7][ctile 0..7][lane 0..31] : one v16bf B-fragment per lane
  __shared__ v16bf sW[8 * 8 * 32];               // 64 KB

  const int tid  = (int)threadIdx.x;
  const int wv   = tid >> 5;
  const int lane = tid & 31;
  const int half = lane >> 4, l16 = lane & 15;
  const int proj = (int)blockIdx.y;              // 0=Q, 1=K, 2=V

  const float* in   = (proj == 0) ? x  : cond;
  const float* W    = (proj == 0) ? Wq : (proj == 1 ? Wk : Wv);
  const float* bias = (proj == 0) ? bq : (proj == 1 ? bk : bv);

  // ---- stage W into LDS, coalesced float4 reads, bf16 fragment scatter ----
  {
    const float4* W4 = (const float4*)W;         // 8192 float4 total
    __bf16* sWh = (__bf16*)sW;
#pragma unroll 1
    for (int i = 0; i < 64; ++i) {
      const int s4   = tid + i * 128;
      const int krow = s4 >> 5;                  // 0..255  (K dim of W)
      const int colb = (s4 & 31) * 4;            // 0..124  (N dim of W)
      const float4 w = W4[s4];
      const int kk  = krow >> 5;                 // 32-deep K chunk
      const int j   = (krow & 7) | (((krow >> 4) & 1) << 3);  // element within fragment
      const int lhi = (krow >> 3) & 1;           // lane-half from K
      const float wf[4] = {w.x, w.y, w.z, w.w};
#pragma unroll
      for (int e = 0; e < 4; ++e) {
        const int col = colb + e;
        const int c   = col >> 4;
        const int ln  = lhi * 16 + (col & 15);
        sWh[((kk * 8 + c) * 32 + ln) * 16 + j] = (__bf16)wf[e];
      }
    }
  }

  float bz[8];
#pragma unroll
  for (int c = 0; c < 8; ++c) bz[c] = bias[c * 16 + l16];

  __syncthreads();

  // ---- 16 row-tiles (16 rows each) per wave against LDS-resident W ----
#pragma unroll 1
  for (int t = 0; t < 16; ++t) {
    const int tile = (int)blockIdx.x * 64 + wv * 16 + t;

    v8f acc[8];
#pragma unroll
    for (int c = 0; c < 8; ++c)
#pragma unroll
      for (int r = 0; r < 8; ++r) acc[c][r] = bz[c];

    const float4* arow4 =
        (const float4*)(in + (size_t)(tile * 16 + l16) * DIN);

#pragma unroll
    for (int kk = 0; kk < 8; ++kk) {
      // A frag 16x32: lane row = l16, elems {half*8+j, 16+half*8+j}
      const float4 f0 = arow4[kk * 8 + half * 2];
      const float4 f1 = arow4[kk * 8 + half * 2 + 1];
      const float4 f2 = arow4[kk * 8 + 4 + half * 2];
      const float4 f3 = arow4[kk * 8 + 4 + half * 2 + 1];
      v16bf a;
      a[0]  = (__bf16)f0.x; a[1]  = (__bf16)f0.y;
      a[2]  = (__bf16)f0.z; a[3]  = (__bf16)f0.w;
      a[4]  = (__bf16)f1.x; a[5]  = (__bf16)f1.y;
      a[6]  = (__bf16)f1.z; a[7]  = (__bf16)f1.w;
      a[8]  = (__bf16)f2.x; a[9]  = (__bf16)f2.y;
      a[10] = (__bf16)f2.z; a[11] = (__bf16)f2.w;
      a[12] = (__bf16)f3.x; a[13] = (__bf16)f3.y;
      a[14] = (__bf16)f3.z; a[15] = (__bf16)f3.w;

#pragma unroll
      for (int c = 0; c < 8; ++c) {
        const v16bf bfr = sW[(kk * 8 + c) * 32 + lane];   // 32B aligned LDS read
        acc[c] = wmma_bf16(a, bfr, acc[c]);
      }
    }

#pragma unroll
    for (int c = 0; c < 8; ++c) {
#pragma unroll
      for (int r = 0; r < 8; ++r) {
        const int row = tile * 16 + half * 8 + r;
        const int col = c * 16 + l16;
        const __bf16 v = (__bf16)acc[c][r];
        if (proj == 0) {
          q_ws[(size_t)row * DOUT + col] = v;
        } else if (proj == 1) {
          k_ws[(size_t)row * DOUT + col] = v;
        } else {
          const int bb = row >> 12;              // row / 4096
          const int rl = row & (LK - 1);
          vt_ws[((size_t)bb * DOUT + col) * LK + rl] = v;   // V transposed
        }
      }
    }
  }
}

// ---------------------------------------------------------------------------
// Kernel 2: flash attention. 4 waves/block, 16 q-rows per wave (64/block).
// Streams 32-key tiles of K and V^T through LDS; online softmax in registers.
// ---------------------------------------------------------------------------
#define KSTR 136   // 128 + 8 halfs pad (bank de-phasing)
#define VSTR 40    // 32 + 8 halfs pad

__global__ __launch_bounds__(128)
void attn(const __bf16* __restrict__ q_ws, const __bf16* __restrict__ k_ws,
          const __bf16* __restrict__ vt_ws, float* __restrict__ out)
{
  __shared__ __bf16 sK[32 * KSTR];       // K tile  [k_local][d]
  __shared__ __bf16 sVt[DOUT * VSTR];    // V tile  [d][k_local]
  __shared__ __bf16 sP[4][16 * 32];      // per-wave P scratch [q_local][k_local]

  const int tid  = (int)threadIdx.x;
  const int wv   = tid >> 5;
  const int lane = tid & 31;
  const int half = lane >> 4;
  const int l16  = lane & 15;

  const int qrow0 = (int)blockIdx.x * 64 + wv * 16;
  const int b     = qrow0 >> 12;

  // Preload all 4 Q A-fragments (16 rows x 128 d) for this wave.
  const __bf16* qrow = q_ws + (size_t)(qrow0 + l16) * DOUT;
  v16bf qa[4];
#pragma unroll
  for (int t = 0; t < 4; ++t) {
    v8bf lo = *(const v8bf*)(qrow + t * 32 + half * 16);
    v8bf hi = *(const v8bf*)(qrow + t * 32 + half * 16 + 8);
    qa[t] = cat8(lo, hi);
  }

  v8f   acc[8];
  float m[8], l[8];
#pragma unroll
  for (int c = 0; c < 8; ++c)
#pragma unroll
    for (int r = 0; r < 8; ++r) acc[c][r] = 0.0f;
#pragma unroll
  for (int r = 0; r < 8; ++r) { m[r] = -3.0e38f; l[r] = 0.0f; }

  const __bf16* kbase = k_ws  + (size_t)b * LK * DOUT;
  const __bf16* vbase = vt_ws + (size_t)b * DOUT * LK;
  __bf16* sPw = &sP[wv][0];

  for (int k32 = 0; k32 < LK; k32 += 32) {
    __syncthreads();   // previous tile fully consumed
    {  // K tile: 32 rows x 128 halfs; 128 threads x 64B each
      const int row = tid >> 2, seg = tid & 3;
      const v8bf* src = (const v8bf*)(kbase + (size_t)(k32 + row) * DOUT + seg * 32);
      v8bf* dst = (v8bf*)(sK + row * KSTR + seg * 32);
      dst[0] = src[0]; dst[1] = src[1]; dst[2] = src[2]; dst[3] = src[3];
    }
    {  // V^T tile: 128 d-rows x 32 halfs; one row per thread
      const v8bf* src = (const v8bf*)(vbase + (size_t)tid * LK + k32);
      v8bf* dst = (v8bf*)(sVt + tid * VSTR);
      dst[0] = src[0]; dst[1] = src[1]; dst[2] = src[2]; dst[3] = src[3];
    }
    __syncthreads();

    // ---- S = Q * K^T  (16q x 32k, f32) ----
    v8f s0 = {0,0,0,0,0,0,0,0}, s1 = {0,0,0,0,0,0,0,0};
#pragma unroll
    for (int t = 0; t < 4; ++t) {
      const __bf16* kp0 = sK + l16 * KSTR + t * 32 + half * 16;
      const __bf16* kp1 = sK + (16 + l16) * KSTR + t * 32 + half * 16;
      v16bf b0 = cat8(*(const v8bf*)kp0, *(const v8bf*)(kp0 + 8));
      v16bf b1 = cat8(*(const v8bf*)kp1, *(const v8bf*)(kp1 + 8));
      s0 = wmma_bf16(qa[t], b0, s0);
      s1 = wmma_bf16(qa[t], b1, s1);
    }

    // ---- online softmax (xor-shuffles stay inside 16-lane half-groups) ----
    float corr[8];
#pragma unroll
    for (int r = 0; r < 8; ++r) {
      float t = fmaxf(s0[r], s1[r]);
#pragma unroll
      for (int off = 1; off < 16; off <<= 1)
        t = fmaxf(t, __shfl_xor(t, off, 32));
      const float mn = fmaxf(m[r], t);
      const float cr = __expf(m[r] - mn);
      m[r] = mn;
      const float p0 = __expf(s0[r] - mn);
      const float p1 = __expf(s1[r] - mn);
      float rs = p0 + p1;
#pragma unroll
      for (int off = 1; off < 16; off <<= 1)
        rs += __shfl_xor(rs, off, 32);
      l[r]    = l[r] * cr + rs;
      corr[r] = cr;
      sPw[(half * 8 + r) * 32 + l16]      = (__bf16)p0;   // P -> LDS (D layout)
      sPw[(half * 8 + r) * 32 + 16 + l16] = (__bf16)p1;
    }
#pragma unroll
    for (int c = 0; c < 8; ++c)
#pragma unroll
      for (int r = 0; r < 8; ++r) acc[c][r] *= corr[r];

    // intra-wave LDS RAW: P stores -> A-layout reads
    asm volatile("s_wait_dscnt 0" ::: "memory");

    const __bf16* pr = sPw + l16 * 32;
    v16bf pa = cat8(*(const v8bf*)(pr + half * 16),
                    *(const v8bf*)(pr + half * 16 + 8));

    // ---- O += P * V  (B-frags contiguous thanks to transposed V) ----
#pragma unroll
    for (int c = 0; c < 8; ++c) {
      const __bf16* vp = sVt + (c * 16 + l16) * VSTR + half * 16;
      v16bf vb = cat8(*(const v8bf*)vp, *(const v8bf*)(vp + 8));
      acc[c] = wmma_bf16(pa, vb, acc[c]);
    }
  }

  // ---- epilogue: O / l, fp32 out ----
  float inv[8];
#pragma unroll
  for (int r = 0; r < 8; ++r) inv[r] = 1.0f / l[r];
#pragma unroll
  for (int c = 0; c < 8; ++c)
#pragma unroll
    for (int r = 0; r < 8; ++r) {
      const int row = qrow0 + half * 8 + r;
      out[(size_t)row * DOUT + c * 16 + l16] = acc[c][r] * inv[r];
    }
}

extern "C" void kernel_launch(void* const* d_in, const int* in_sizes, int n_in,
                              void* d_out, int out_size, void* d_ws, size_t ws_size,
                              hipStream_t stream) {
  const float* x    = (const float*)d_in[0];
  const float* cond = (const float*)d_in[1];
  const float* Wq   = (const float*)d_in[2];
  const float* bq   = (const float*)d_in[3];
  const float* Wk   = (const float*)d_in[4];
  const float* bk   = (const float*)d_in[5];
  const float* Wv   = (const float*)d_in[6];
  const float* bv   = (const float*)d_in[7];

  __bf16* q_ws  = (__bf16*)d_ws;                         // 4 MB
  __bf16* k_ws  = q_ws + (size_t)BB * LQ * DOUT;         // 4 MB
  __bf16* vt_ws = k_ws + (size_t)BB * LK * DOUT;         // 4 MB (transposed V)

  // 3 projections x 16 tile-groups (64 row-tiles per block)
  dim3 pgrid(16, 3);
  proj_qkv<<<pgrid, 128, 0, stream>>>(x, cond, Wq, bq, Wk, bk, Wv, bv,
                                      q_ws, k_ws, vt_ws);
  // 16384 q-rows / 64 per block
  attn<<<(BB * LQ) / 64, 128, 0, stream>>>(q_ws, k_ws, vt_ws, (float*)d_out);
}
